// Gate_25984552141449
// MI455X (gfx1250) — compile-verified, matrix operand order checked by reading, not run on hardware
//
#include <hip/hip_runtime.h>
#include <hip/hip_bf16.h>

#define DIM     4096
#define NEXP    256
#define TOKENS  8192
#define BLK_M   64
#define TOPK    8
#define NGRP    8
#define TOPG    4
#define CH      64                  // K per chunk
#define NCHUNK  (DIM / CH)          // 64 chunks

// unified padded LDS row: 64 bf16 = 128B data + 16B pad
#define ROWB    144
#define A_BUFB  (BLK_M * ROWB)      // 9216 B per buffer (bf16 A tile)
#define B_BUFB  (NEXP  * ROWB)      // 36864 B per buffer (bf16 B tile)
#define SMEM_BYTES (2 * A_BUFB + 2 * B_BUFB)   // 92160 B (>= 64KB score tile)

typedef __bf16  v16bf __attribute__((ext_vector_type(16)));
typedef __bf16  v8bf  __attribute__((ext_vector_type(8)));
typedef __bf16  v4bf  __attribute__((ext_vector_type(4)));
typedef float   v8f   __attribute__((ext_vector_type(8)));
typedef float   f4    __attribute__((ext_vector_type(4)));

__device__ __forceinline__ float bperm_f(int srcLane, float v) {
    return __int_as_float(__builtin_amdgcn_ds_bpermute(srcLane << 2, __float_as_int(v)));
}
__device__ __forceinline__ int bperm_i(int srcLane, int v) {
    return __builtin_amdgcn_ds_bpermute(srcLane << 2, v);
}
__device__ __forceinline__ void async_b128(unsigned lds_addr, const void* gaddr) {
    asm volatile("global_load_async_to_lds_b128 %0, %1, off"
                 :: "v"(lds_addr), "v"(gaddr) : "memory");
}
__device__ __forceinline__ void wait_async0() {
    asm volatile("s_wait_asynccnt 0x0" ::: "memory");
}
__device__ __forceinline__ unsigned lds_lo32(const void* p) {
    return (unsigned)(unsigned long long)p;   // low 32 bits of generic ptr = LDS offset
}

// load one 16x32 bf16 fragment row-slice: 16 bf16 (32B) per lane, 2x ds_load_b128
__device__ __forceinline__ v16bf load_frag(const char* base, int n, int h, int lhi) {
    const v8bf* p = (const v8bf*)(base + n * ROWB + h * 64 + lhi * 32);
    v8bf b0 = p[0], b1 = p[1];
    v16bf r;
#pragma unroll
    for (int q = 0; q < 8; ++q) { r[q] = b0[q]; r[8 + q] = b1[q]; }
    return r;
}

// ---------------- pre-pass: W f32 -> bf16 ------------------------------------
__global__ __launch_bounds__(256) void
convert_w_kernel(const float* __restrict__ w, __bf16* __restrict__ o)
{
    int i = blockIdx.x * blockDim.x + threadIdx.x;
    const f4* p = (const f4*)w;
    f4 v = p[i];
    v4bf r;
#pragma unroll
    for (int j = 0; j < 4; ++j) r[j] = (__bf16)v[j];
    *(v4bf*)(o + (size_t)i * 4) = r;
}

// ---------------- fused gate GEMM (bf16 WMMA) + sigmoid + routing ------------
__global__ __launch_bounds__(256) void
gate_router_kernel(const float* __restrict__ x,
                   const __bf16* __restrict__ wbf,
                   const float* __restrict__ bias,
                   float* __restrict__ out_w,
                   int*   __restrict__ out_idx)
{
    extern __shared__ char smem[];
    float* sc = (float*)smem;                 // [64][256], live AFTER K-loop only

    const int tid  = threadIdx.x;
    const int lane = tid & 31;
    const int wave = tid >> 5;
    const int wm2  = wave & 1;                // M half (0..1): 32 rows
    const int wn4  = wave >> 1;               // N quarter (0..3): 64 experts
    const int l16  = lane & 15;
    const int lhi  = lane >> 4;
    const int m0   = blockIdx.x * BLK_M;
    const int nb   = wn4 * 64;

    // ---- A staging (reg pipeline, f32 -> bf16): 64 rows x 256B, 64B/thread ----
    const char* gA = (const char*)(x + (size_t)(m0 + (tid >> 2)) * DIM) + (tid & 3) * 64;
    const unsigned offAst = (unsigned)(tid >> 2) * ROWB + (tid & 3) * 32;
    // ---- B staging (async): e = tid + j*256 -> row (tid>>3)+j*32, part tid&7 --
    const char* gB = (const char*)(wbf + (size_t)(tid >> 3) * DIM) + (tid & 7) * 16;
    const unsigned offB0 = (unsigned)(tid >> 3) * ROWB + (tid & 7) * 16;

    const unsigned ldsBase = lds_lo32(smem);
    const unsigned bBase[2] = { ldsBase + 2 * A_BUFB, ldsBase + 2 * A_BUFB + B_BUFB };

    f4 pa[4];   // A prefetch registers (16 f32)
    v8f acc[8]; // acc[mt*4+nt]
#pragma unroll
    for (int t = 0; t < 8; ++t) acc[t] = (v8f){0.f,0.f,0.f,0.f,0.f,0.f,0.f,0.f};

    // prologue: A chunk 0 via regs, B chunk 0 via async, both into buffer 0
#pragma unroll
    for (int q = 0; q < 4; ++q) pa[q] = ((const f4*)gA)[q];
    gA += CH * 4;
#pragma unroll
    for (int j = 0; j < 8; ++j) async_b128(bBase[0] + offB0 + j * 4608u, gB + (size_t)j * 262144);
    gB += CH * 2;
    {
        v8bf c0, c1;
#pragma unroll
        for (int q = 0; q < 4; ++q) {
            c0[q] = (__bf16)pa[0][q]; c0[4 + q] = (__bf16)pa[1][q];
            c1[q] = (__bf16)pa[2][q]; c1[4 + q] = (__bf16)pa[3][q];
        }
        char* d = smem + offAst;
        *(v8bf*)d = c0; *(v8bf*)(d + 16) = c1;
    }
    wait_async0();
    __syncthreads();

    for (int i = 0; i < NCHUNK; ++i) {
        const int cur = i & 1;
        const int nxt = cur ^ 1;
        const bool more = (i + 1 < NCHUNK);
        if (more) {   // prefetch: A f32 into regs, B async into other buffer
#pragma unroll
            for (int q = 0; q < 4; ++q) pa[q] = ((const f4*)gA)[q];
            gA += CH * 4;
#pragma unroll
            for (int j = 0; j < 8; ++j) async_b128(bBase[nxt] + offB0 + j * 4608u, gB + (size_t)j * 262144);
            gB += CH * 2;
        }

        const char* sAc = smem + cur * A_BUFB;
        const char* sBc = smem + 2 * A_BUFB + cur * B_BUFB;

        // ---- four A fragments (bf16, direct): M tiles x K halves --------------
        v16bf afr[4];   // afr[mt*2+h]
#pragma unroll
        for (int mt = 0; mt < 2; ++mt)
#pragma unroll
            for (int h = 0; h < 2; ++h)
                afr[mt * 2 + h] = load_frag(sAc, wm2 * 32 + mt * 16 + l16, h, lhi);

        // ---- 4 N-tiles; each B fragment pair feeds 4 WMMAs --------------------
        v16bf b0 = load_frag(sBc, nb + l16, 0, lhi);
        v16bf b1 = load_frag(sBc, nb + l16, 1, lhi);
#pragma unroll
        for (int nt = 0; nt < 4; ++nt) {
            v16bf cb0 = b0, cb1 = b1;
            if (nt < 3) {
                b0 = load_frag(sBc, nb + (nt + 1) * 16 + l16, 0, lhi);
                b1 = load_frag(sBc, nb + (nt + 1) * 16 + l16, 1, lhi);
            }
            acc[nt] = __builtin_amdgcn_wmma_f32_16x16x32_bf16(
                false, afr[0], false, cb0, (short)0, acc[nt], false, false);
            acc[nt] = __builtin_amdgcn_wmma_f32_16x16x32_bf16(
                false, afr[1], false, cb1, (short)0, acc[nt], false, false);
            acc[4 + nt] = __builtin_amdgcn_wmma_f32_16x16x32_bf16(
                false, afr[2], false, cb0, (short)0, acc[4 + nt], false, false);
            acc[4 + nt] = __builtin_amdgcn_wmma_f32_16x16x32_bf16(
                false, afr[3], false, cb1, (short)0, acc[4 + nt], false, false);
        }

        if (more) {   // convert prefetched A and store into next buffer
            v8bf c0, c1;
#pragma unroll
            for (int q = 0; q < 4; ++q) {
                c0[q] = (__bf16)pa[0][q]; c0[4 + q] = (__bf16)pa[1][q];
                c1[q] = (__bf16)pa[2][q]; c1[4 + q] = (__bf16)pa[3][q];
            }
            char* d = smem + nxt * A_BUFB + offAst;
            *(v8bf*)d = c0; *(v8bf*)(d + 16) = c1;
        }

        wait_async0();       // own next-chunk B copies done
        __syncthreads();     // all waves done with buf[cur]; stores/copies visible
    }

    // ---- sigmoid + scatter into LDS score tile (reuses GEMM LDS) --------------
#pragma unroll
    for (int mt = 0; mt < 2; ++mt) {
#pragma unroll
        for (int nt = 0; nt < 4; ++nt) {
#pragma unroll
            for (int r = 0; r < 8; ++r) {
                float z = acc[mt * 4 + nt][r];
                float s = 1.0f / (1.0f + __expf(-z));
                int row = wm2 * 32 + mt * 16 + r + lhi * 8;
                int col = nb + nt * 16 + l16;
                sc[row * NEXP + col] = s;
            }
        }
    }
    __syncthreads();

    // ================= routing: 8 tokens per wave, 8 experts per lane ==========
    const float NEG = -__builtin_inff();
    float b8[8];
#pragma unroll
    for (int j = 0; j < 8; ++j) b8[j] = bias[lane * 8 + j];

    for (int i = 0; i < 8; ++i) {
        const int trow = wave * 8 + i;
        const int tok  = blockIdx.x * BLK_M + trow;

        float v[8];
#pragma unroll
        for (int j = 0; j < 8; ++j)
            v[j] = sc[trow * NEXP + lane * 8 + j] + b8[j];

        float t1 = NEG, t2 = NEG;
#pragma unroll
        for (int j = 0; j < 8; ++j) {
            float a = v[j];
            if (a > t1) { t2 = t1; t1 = a; } else if (a > t2) { t2 = a; }
        }
#pragma unroll
        for (int m = 1; m <= 2; m <<= 1) {
            float o1 = bperm_f(lane ^ m, t1);
            float o2 = bperm_f(lane ^ m, t2);
            float hi = fmaxf(t1, o1);
            float lo = fmaxf(fminf(t1, o1), fmaxf(t2, o2));
            t1 = hi; t2 = lo;
        }
        float gs = t1 + t2;

        float g[NGRP];
#pragma unroll
        for (int q = 0; q < NGRP; ++q) g[q] = bperm_f(q * 4, gs);

        unsigned keep = 0;
#pragma unroll
        for (int r = 0; r < TOPG; ++r) {
            int best = 0; float bv = NEG;
#pragma unroll
            for (int q = 0; q < NGRP; ++q)
                if (!((keep >> q) & 1) && g[q] > bv) { bv = g[q]; best = q; }
            keep |= 1u << best;
        }
        if (!((keep >> (lane >> 2)) & 1)) {
#pragma unroll
            for (int j = 0; j < 8; ++j) v[j] = NEG;
        }

        float myW = 0.f; int myIdx = 0; float wsum = 0.f;
        for (int k = 0; k < TOPK; ++k) {
            float lv = NEG; int lj = 0;
#pragma unroll
            for (int j = 0; j < 8; ++j)
                if (v[j] > lv) { lv = v[j]; lj = j; }
            float rv = lv; int re = lane * 8 + lj;
#pragma unroll
            for (int m = 1; m < 32; m <<= 1) {
                float ov = bperm_f(lane ^ m, rv);
                int   oe = bperm_i(lane ^ m, re);
                if (ov > rv || (ov == rv && oe < re)) { rv = ov; re = oe; }
            }
            if ((re >> 3) == lane) v[re & 7] = NEG;
            float wk = sc[trow * NEXP + re];      // ORIGINAL (un-biased) score
            wsum += wk;
            if (lane == k) { myW = wk; myIdx = re; }
        }
        if (lane < TOPK) {
            out_w[(size_t)tok * TOPK + lane]   = myW / wsum * 2.5f;
            out_idx[(size_t)tok * TOPK + lane] = myIdx;
        }
    }
}

extern "C" void kernel_launch(void* const* d_in, const int* in_sizes, int n_in,
                              void* d_out, int out_size, void* d_ws, size_t ws_size,
                              hipStream_t stream) {
    (void)in_sizes; (void)n_in; (void)out_size; (void)ws_size;
    const float* x    = (const float*)d_in[0];
    const float* wgt  = (const float*)d_in[1];
    const float* bias = (const float*)d_in[2];
    float* out_w  = (float*)d_out;                       // [TOKENS, 8] f32
    int*   out_id = (int*)(out_w + (size_t)TOKENS * 8);  // [TOKENS, 8] i32
    __bf16* wbf   = (__bf16*)d_ws;                       // 2 MB bf16 weights

    convert_w_kernel<<<dim3((NEXP * DIM) / 4 / 256), dim3(256), 0, stream>>>(wgt, wbf);

    dim3 grid(TOKENS / BLK_M);   // 128 blocks
    dim3 block(256);             // 8 waves (wave32)
    gate_router_kernel<<<grid, block, SMEM_BYTES, stream>>>(x, wbf, bias, out_w, out_id);
}
